// RNN_63110249448069
// MI455X (gfx1250) — matrix operand order repaired and checked by reading, not compiled, not run on hardware
//
#include <hip/hip_runtime.h>
#include <stdint.h>

// Problem constants
constexpr int BATCH = 64;
constexpr int SEQ   = 512;
constexpr int IN_F  = 512;
constexpr int HID   = 1024;
constexpr int OUT_F = 512;

typedef __attribute__((ext_vector_type(16))) __bf16 v16bf;
typedef __attribute__((ext_vector_type(8)))  float  v8f;

union BF16Frag {
    v16bf v;
    uint4 q[2];
    unsigned short u[16];
};

__device__ __forceinline__ unsigned short f2bf(float f) {
    unsigned int x = __float_as_uint(f);
    unsigned int r = x + 0x7FFFu + ((x >> 16) & 1u);   // round-to-nearest-even
    return (unsigned short)(r >> 16);
}

// ---------------------------------------------------------------------------
// Kernel 1: vectorized f32 -> bf16 conversion (8 elements / thread, b128 I/O)
// ---------------------------------------------------------------------------
__global__ __launch_bounds__(256)
void cvt8_f32_bf16_kernel(const float4* __restrict__ in,
                          uint4* __restrict__ out, int n8) {
    int i = blockIdx.x * blockDim.x + threadIdx.x;
    if (i < n8) {
        float4 a = in[2 * i];
        float4 b = in[2 * i + 1];
        uint4 o;
        o.x = (unsigned)f2bf(a.x) | ((unsigned)f2bf(a.y) << 16);
        o.y = (unsigned)f2bf(a.z) | ((unsigned)f2bf(a.w) << 16);
        o.z = (unsigned)f2bf(b.x) | ((unsigned)f2bf(b.y) << 16);
        o.w = (unsigned)f2bf(b.z) | ((unsigned)f2bf(b.w) << 16);
        out[i] = o;
    }
}

// ---------------------------------------------------------------------------
// Kernel 2: xp[t][h][b] = x[b][t][:] . Wi[h][:] + bi[h]
// xb is bf16 x viewed as [B*S, IN] row-major; flat row g -> b = g>>9, t = g&511.
// Block: 256 threads = 8 waves; wave does one 16x16 tile, K = 512.
// All WMMA operands load straight from memory as b128 (no cvt in the loop).
// ---------------------------------------------------------------------------
__global__ __launch_bounds__(256)
void xproj_kernel(const unsigned short* __restrict__ xb,     // [B*S][IN] bf16
                  const unsigned short* __restrict__ Wib,    // [HID][IN] bf16
                  const float* __restrict__ bi,
                  float* __restrict__ xp) {                  // [SEQ][HID][BATCH]
    const int lane  = threadIdx.x & 31;
    const int wave  = threadIdx.x >> 5;
    const int m0    = blockIdx.x * 16;                  // row tile of x-flat
    const int n0    = blockIdx.y * 128 + wave * 16;     // hid tile

    const int mA    = lane & 15;            // A: row within tile
    const int kA    = (lane >> 4) * 8;      // A: K sub-chunk base
    const int kB    = (lane >> 4) * 8;      // B: K sub-chunk base
    const int nC    = lane & 15;            // C/D: column
    const int mHalf = (lane >> 4) * 8;      // C/D: row half base

    const unsigned short* xrow = xb  + (size_t)(m0 + mA) * IN_F + kA;
    const unsigned short* wrow = Wib + (size_t)(n0 + nC) * IN_F + kB;

    const float bias = bi[n0 + nC];
    v8f acc;
#pragma unroll
    for (int r = 0; r < 8; ++r) acc[r] = bias;

    for (int kb = 0; kb < IN_F; kb += 32) {
        BF16Frag a, b;
        a.q[0] = *(const uint4*)(xrow + kb);
        a.q[1] = *(const uint4*)(xrow + kb + 16);
        b.q[0] = *(const uint4*)(wrow + kb);
        b.q[1] = *(const uint4*)(wrow + kb + 16);
        acc = __builtin_amdgcn_wmma_f32_16x16x32_bf16(
                  false, a.v, false, b.v, (short)0, acc, false, false);
    }

    // D element (m, n) -> xp[t][n][b] with g = m0 + m, b = g>>9, t = g&511
#pragma unroll
    for (int r = 0; r < 8; ++r) {
        int g  = m0 + mHalf + r;
        int bb = g >> 9;                 // / SEQ
        int tt = g & (SEQ - 1);          // % SEQ
        xp[((size_t)tt * HID + n0 + nC) * BATCH + bb] = acc[r];
    }
}

// ---------------------------------------------------------------------------
// Kernel 3: persistent per-batch-group scan. Grid = BATCH/16 = 4 workgroups,
// 512 threads = 16 wave32s. h[16][HID] lives in LDS as bf16 for all 512 steps.
// ---------------------------------------------------------------------------
constexpr int H_STRIDE = HID + 8;   // 1032 bf16 -> 2064B row stride (16B-aligned,
                                    // breaks 64-bank conflicts on ds_read_b128)

__global__ __launch_bounds__(512)
void rnn_scan_kernel(const float* __restrict__ xp,            // [SEQ][HID][BATCH]
                     const unsigned short* __restrict__ Whb,  // [HID][HID] bf16
                     const float* __restrict__ bh,
                     const unsigned short* __restrict__ Wob,  // [OUT][HID] bf16
                     const float* __restrict__ bo,
                     const float* __restrict__ ln_g,
                     const float* __restrict__ ln_b,
                     float* __restrict__ out) {               // [SEQ][BATCH][OUT]
    __shared__ unsigned short hsh[16 * H_STRIDE];  // ~33 KB
    __shared__ float part_s[16][16];
    __shared__ float part_q[16][16];
    __shared__ float mu_sh[16];
    __shared__ float rs_sh[16];

    const int lane  = threadIdx.x & 31;
    const int wave  = threadIdx.x >> 5;     // 0..15
    const int b0    = blockIdx.x * 16;      // batch rows owned by this WG

    const int mA    = lane & 15;
    const int kA    = (lane >> 4) * 8;
    const int kB    = (lane >> 4) * 8;
    const int nC    = lane & 15;
    const int mHalf = (lane >> 4) * 8;

    // h0 = 0
    for (int i = threadIdx.x; i < 16 * H_STRIDE; i += blockDim.x) hsh[i] = 0;
    __syncthreads();

    // Per-wave invariants for the inner GEMM (4 tiles) and output GEMM (2 tiles)
    float bh4[4], g4[4], lb4[4];
    const unsigned short* whrow[4];
#pragma unroll
    for (int i = 0; i < 4; ++i) {
        int n = (wave * 4 + i) * 16 + nC;
        bh4[i] = bh[n];
        g4[i]  = ln_g[n];
        lb4[i] = ln_b[n];
        whrow[i] = Whb + (size_t)n * HID + kB;
    }
    float bo2[2];
    const unsigned short* worow[2];
#pragma unroll
    for (int i = 0; i < 2; ++i) {
        int n = (wave * 2 + i) * 16 + nC;
        bo2[i] = bo[n];
        worow[i] = Wob + (size_t)n * HID + kB;
    }

    for (int t = 0; t < SEQ; ++t) {
        // ---- Phase 1: inner = h @ Wh^T + bh + xp_t  (4 N-tiles per wave) ----
        // Seed: xp is [t][h][b], so the 8 D-rows of a tile are 8 consecutive
        // floats -> two b128 loads per tile.
        v8f acc[4];
#pragma unroll
        for (int i = 0; i < 4; ++i) {
            const float* sp = xp + ((size_t)t * HID + (wave * 4 + i) * 16 + nC) * BATCH
                                 + b0 + mHalf;
            float4 s0 = *(const float4*)(sp);
            float4 s1 = *(const float4*)(sp + 4);
            acc[i][0] = bh4[i] + s0.x;  acc[i][1] = bh4[i] + s0.y;
            acc[i][2] = bh4[i] + s0.z;  acc[i][3] = bh4[i] + s0.w;
            acc[i][4] = bh4[i] + s1.x;  acc[i][5] = bh4[i] + s1.y;
            acc[i][6] = bh4[i] + s1.z;  acc[i][7] = bh4[i] + s1.w;
        }
        for (int kb = 0; kb < HID; kb += 32) {
            BF16Frag a;
            a.q[0] = *(const uint4*)(hsh + mA * H_STRIDE + kb + kA);
            a.q[1] = *(const uint4*)(hsh + mA * H_STRIDE + kb + kA + 16);
#pragma unroll
            for (int i = 0; i < 4; ++i) {
                BF16Frag b;
                b.q[0] = *(const uint4*)(whrow[i] + kb);
                b.q[1] = *(const uint4*)(whrow[i] + kb + 16);
                acc[i] = __builtin_amdgcn_wmma_f32_16x16x32_bf16(
                             false, a.v, false, b.v, (short)0, acc[i], false, false);
            }
        }

        // ---- Phase 2: LayerNorm + tanh, new h stored to LDS as bf16 ----
#pragma unroll
        for (int r = 0; r < 8; ++r) {
            float s = acc[0][r] + acc[1][r] + acc[2][r] + acc[3][r];
            float q = acc[0][r] * acc[0][r] + acc[1][r] * acc[1][r]
                    + acc[2][r] * acc[2][r] + acc[3][r] * acc[3][r];
#pragma unroll
            for (int off = 1; off < 16; off <<= 1) {
                s += __shfl_xor(s, off, 32);
                q += __shfl_xor(q, off, 32);
            }
            if (nC == 0) {   // lanes 0 and 16 -> rows mHalf+r
                part_s[wave][mHalf + r] = s;
                part_q[wave][mHalf + r] = q;
            }
        }
        __syncthreads();
        if (threadIdx.x < 16) {
            float s = 0.f, q = 0.f;
#pragma unroll
            for (int w = 0; w < 16; ++w) { s += part_s[w][threadIdx.x];
                                           q += part_q[w][threadIdx.x]; }
            float mu  = s * (1.0f / HID);
            float var = q * (1.0f / HID) - mu * mu;
            mu_sh[threadIdx.x] = mu;
            rs_sh[threadIdx.x] = rsqrtf(var + 1e-5f);
        }
        __syncthreads();
#pragma unroll
        for (int i = 0; i < 4; ++i) {
            int n = (wave * 4 + i) * 16 + nC;
#pragma unroll
            for (int r = 0; r < 8; ++r) {
                int m = mHalf + r;
                float v = (acc[i][r] - mu_sh[m]) * rs_sh[m] * g4[i] + lb4[i];
                hsh[m * H_STRIDE + n] = f2bf(tanhf(v));
            }
        }
        __syncthreads();

        // ---- Phase 3: y = h_new @ Wo^T + bo  (2 N-tiles per wave) ----
#pragma unroll
        for (int i = 0; i < 2; ++i) {
            v8f accy;
#pragma unroll
            for (int r = 0; r < 8; ++r) accy[r] = bo2[i];
            for (int kb = 0; kb < HID; kb += 32) {
                BF16Frag a, b;
                a.q[0] = *(const uint4*)(hsh + mA * H_STRIDE + kb + kA);
                a.q[1] = *(const uint4*)(hsh + mA * H_STRIDE + kb + kA + 16);
                b.q[0] = *(const uint4*)(worow[i] + kb);
                b.q[1] = *(const uint4*)(worow[i] + kb + 16);
                accy = __builtin_amdgcn_wmma_f32_16x16x32_bf16(
                           false, a.v, false, b.v, (short)0, accy, false, false);
            }
            int n = (wave * 2 + i) * 16 + nC;
#pragma unroll
            for (int r = 0; r < 8; ++r)
                out[((size_t)t * BATCH + b0 + mHalf + r) * OUT_F + n] = accy[r];
        }
        __syncthreads();   // protect hsh before next step's Phase 1 reads
    }
}

// ---------------------------------------------------------------------------
// Host launch
// ---------------------------------------------------------------------------
extern "C" void kernel_launch(void* const* d_in, const int* in_sizes, int n_in,
                              void* d_out, int out_size, void* d_ws, size_t ws_size,
                              hipStream_t stream) {
    (void)in_sizes; (void)n_in; (void)out_size; (void)ws_size;
    const float* x   = (const float*)d_in[0];   // [B,S,IN]
    const float* Wh  = (const float*)d_in[1];   // [HID,HID]
    const float* bh  = (const float*)d_in[2];
    const float* Wi  = (const float*)d_in[3];   // [HID,IN]
    const float* bi  = (const float*)d_in[4];
    const float* Wo  = (const float*)d_in[5];   // [OUT,HID]
    const float* bo  = (const float*)d_in[6];
    const float* lng = (const float*)d_in[7];
    const float* lnb = (const float*)d_in[8];
    float* out = (float*)d_out;

    // Workspace carve-up (all offsets >=256B aligned):
    //   [0,   2MiB)   Whb bf16 [HID][HID]
    //   [2,   3MiB)   Wib bf16 [HID][IN]
    //   [3,   4MiB)   Wob bf16 [OUT][HID]
    //   [4, 132MiB)   xp  f32  [SEQ][HID][BATCH]
    //   [132,164MiB)  xb  bf16 [B*S][IN]
    unsigned char* ws = (unsigned char*)d_ws;
    unsigned short* Whb = (unsigned short*)(ws);
    unsigned short* Wib = (unsigned short*)(ws + (2llu << 20));
    unsigned short* Wob = (unsigned short*)(ws + (3llu << 20));
    float*          xp  = (float*)        (ws + (4llu << 20));
    unsigned short* xb  = (unsigned short*)(ws + (132llu << 20));

    const int nWh = HID * HID, nWi = HID * IN_F, nWo = OUT_F * HID;
    const int nX  = BATCH * SEQ * IN_F;
    cvt8_f32_bf16_kernel<<<(nWh / 8 + 255) / 256, 256, 0, stream>>>(
        (const float4*)Wh, (uint4*)Whb, nWh / 8);
    cvt8_f32_bf16_kernel<<<(nWi / 8 + 255) / 256, 256, 0, stream>>>(
        (const float4*)Wi, (uint4*)Wib, nWi / 8);
    cvt8_f32_bf16_kernel<<<(nWo / 8 + 255) / 256, 256, 0, stream>>>(
        (const float4*)Wo, (uint4*)Wob, nWo / 8);
    cvt8_f32_bf16_kernel<<<(nX / 8 + 255) / 256, 256, 0, stream>>>(
        (const float4*)x, (uint4*)xb, nX / 8);

    dim3 gX((BATCH * SEQ) / 16, HID / 128);     // 2048 x 8 workgroups
    xproj_kernel<<<gX, 256, 0, stream>>>(xb, Wib, bi, xp);

    rnn_scan_kernel<<<BATCH / 16, 512, 0, stream>>>(xp, Whb, bh, Wob, bo,
                                                    lng, lnb, out);
}
